// InputsSchedulePredictorNN_11055245820208
// MI455X (gfx1250) — matrix operand order repaired and checked by reading, not compiled, not run on hardware
//
#include <hip/hip_runtime.h>
#include <math.h>

#define DEVINL __device__ __forceinline__

typedef _Float16 v16h __attribute__((ext_vector_type(16)));
typedef _Float16 v8h  __attribute__((ext_vector_type(8)));
typedef float    v8f  __attribute__((ext_vector_type(8)));

constexpr int Bsz = 16384, T = 50, INF = 5, H = 64;
constexpr int PE1 = 32, PE2 = 16, PD1 = 64, PD2 = 16, OUTF = 2, TS = 35;
constexpr float DTc = 0.033f, VEL_SCALING = 0.1f;
constexpr float JERK_LIM = 1.0f, STEER_RATE_LIM = 1.0f;

constexpr int WAVES = 4;                   // waves per block
constexpr int BLOCK = WAVES * 32;          // 128 threads
constexpr int ROWS_PER_BLOCK = WAVES * 16; // 64 batch rows per block
constexpr int KP = 96;                     // padded K stride (3 x 32 WMMA chunks)

// Unified K layout (cols of A and of every weight matrix, zero-padded):
//   [0..3]   : decoder LSTM input (head,tail)  / zero in encoder
//   [4..67]  : hidden state h (W_hh columns)
//   [68..83] : encoder PE output (enc W_ih)    / zero in decoder
//   [84..95] : zero
struct __align__(64) SMem {
  _Float16 W  [256 * KP];          // 48 KB combined gate weights (enc, then dec)
  _Float16 PDW[PD1 * KP];          // 12 KB projection weights (dec phase)
  _Float16 FW1[PD2 * KP];          //  3 KB final layer-1 weights
  _Float16 A  [WAVES * 16 * KP];   // 12 KB per-wave activation staging (WMMA A)
  _Float16 A2 [WAVES * 16 * KP];   // 12 KB per-wave staging for proj->fin1
  float h1buf[WAVES * 16 * PE1];   //  8 KB PE hidden scratch
  float f1buf[WAVES * 16 * PD2];   //  4 KB fin1 output scratch
  float bias[256];
  float pdb[PD1];
  float f1b[PD2];
  float fw2[OUTF * PD2];
  float f2b[OUTF];
  float pw1[PE1 * INF];
  float pb1[PE1];
  float pw2[PE2 * PE1];
  float pb2[PE2];
  int   dsx[ROWS_PER_BLOCK];
};

DEVINL void wave_sync() {
  // order this wave's LDS stores before subsequent cross-lane LDS reads
  __builtin_amdgcn_wave_barrier();
  asm volatile("s_wait_dscnt 0" ::: "memory");
  __builtin_amdgcn_wave_barrier();
}

// Hardware tanh (gfx1250 V_TANH_F32 trans op; co-executes with WMMA).
DEVINL float htanh(float x) {
#if __has_builtin(__builtin_amdgcn_tanhf)
  return __builtin_amdgcn_tanhf(x);
#elif __has_builtin(__builtin_amdgcn_tanh_f32)
  return __builtin_amdgcn_tanh_f32(x);
#else
  float r;
  asm("v_tanh_f32 %0, %1\n\tv_nop" : "=v"(r) : "v"(x));
  return r;
#endif
}

// sigmoid(x + b) = 0.5 * tanh(0.5*x + 0.5*b) + 0.5   (hb = 0.5*b precomputed)
DEVINL float sigfb(float x, float hb) {
  return fmaf(0.5f, htanh(fmaf(0.5f, x, hb)), 0.5f);
}

// A fragment (16x32 f16, M = batch rows): lane holds 16 contiguous K of row
// M = lane%16, starting at K = kc*32 + (lane/16)*16.
DEVINL v16h load_a(const _Float16* A, int lane, int kc) {
  const int m  = lane & 15;
  const int k0 = kc * 32 + (lane >> 4) * 16;
  return *(const v16h*)(A + m * KP + k0);
}

// B fragment (32x16 f16, N = output cols): lane holds 16 contiguous K of column
// n = nbase + lane%16, starting at K = kc*32 + (lane/16)*16. W is [N][KP] row-major.
DEVINL v16h load_b(const _Float16* W, int lane, int nbase, int kc) {
  const int n  = nbase + (lane & 15);
  const int k0 = kc * 32 + (lane >> 4) * 16;
  return *(const v16h*)(W + n * KP + k0);
}

DEVINL v8f wmma_acc(v16h a, v16h b, v8f c) {
  return __builtin_amdgcn_wmma_f32_16x16x32_f16(false, a, false, b, (short)0, c, false, false);
}

// 4 output tiles (cols nb[g]..nb[g]+15) over 3 K-chunks with double-buffered
// B fragments: loads for chunk kc+1 are issued before the 4 WMMAs of chunk kc,
// so ds_load_b128s overlap the matrix pipe (partial s_wait_dscnt).
DEVINL void mm4x3_pipe(const _Float16* W, const int* nb, const v16h* a,
                       v8f* acc, int lane) {
  v16h bb[2][4];
#pragma unroll
  for (int g = 0; g < 4; ++g) bb[0][g] = load_b(W, lane, nb[g], 0);
#pragma unroll
  for (int kc = 0; kc < 3; ++kc) {
    if (kc < 2) {
#pragma unroll
      for (int g = 0; g < 4; ++g)
        bb[(kc + 1) & 1][g] = load_b(W, lane, nb[g], kc + 1);
    }
#pragma unroll
    for (int g = 0; g < 4; ++g) {
      if (kc == 0) {
        v8f z = {0.f, 0.f, 0.f, 0.f, 0.f, 0.f, 0.f, 0.f};
        acc[g] = wmma_acc(a[0], bb[0][g], z);
      } else {
        acc[g] = wmma_acc(a[kc], bb[kc & 1][g], acc[g]);
      }
    }
  }
}

// One LSTM step: gates = A * W^T, elementwise update of c (registers) and
// h (written back as f16 into A cols [4..67]). Biases folded into epilogue.
DEVINL void lstm_gates(const _Float16* W, const float* bias, _Float16* Aw,
                       float* creg, int lane) {
  const int ln16 = lane & 15, hs = lane >> 4;
  v16h a[3];
  a[0] = load_a(Aw, lane, 0);
  a[1] = load_a(Aw, lane, 1);
  a[2] = load_a(Aw, lane, 2);
#pragma unroll
  for (int j = 0; j < 4; ++j) {
    const int nb[4] = {0 + 16 * j, 64 + 16 * j, 128 + 16 * j, 192 + 16 * j};
    v8f acc[4];  // i, f, g, o gate tiles
    mm4x3_pipe(W, nb, a, acc, lane);
    const float hbI = 0.5f * bias[nb[0]];  // note: bias col = nb[g] + ln16
    const float hbF = 0.5f * bias[nb[1] + ln16];
    const float bG  =        bias[nb[2] + ln16];
    const float hbO = 0.5f * bias[nb[3] + ln16];
    const float hbI2 = 0.5f * bias[nb[0] + ln16];
    (void)hbI;
#pragma unroll
    for (int v = 0; v < 8; ++v) {
      float it = sigfb(acc[0][v], hbI2);
      float ft = sigfb(acc[1][v], hbF);
      float ot = sigfb(acc[3][v], hbO);
      float gt = htanh(acc[2][v] + bG);
      float cn = fmaf(ft, creg[j * 8 + v], it * gt);
      float hn = ot * htanh(cn);
      creg[j * 8 + v] = cn;
      Aw[(v + 8 * hs) * KP + 4 + 16 * j + ln16] = (_Float16)hn;
    }
  }
}

__global__ __launch_bounds__(BLOCK) void predictor_kernel(
    const float* __restrict__ x, const int* __restrict__ dataset_idx,
    const float* __restrict__ pe_w1, const float* __restrict__ pe_b1,
    const float* __restrict__ pe_w2, const float* __restrict__ pe_b2,
    const float* __restrict__ enc_wih, const float* __restrict__ enc_whh,
    const float* __restrict__ enc_bih, const float* __restrict__ enc_bhh,
    const float* __restrict__ dec_wih, const float* __restrict__ dec_whh,
    const float* __restrict__ dec_bih, const float* __restrict__ dec_bhh,
    const float* __restrict__ pd_w, const float* __restrict__ pd_b,
    const float* __restrict__ scales, const float* __restrict__ fin_w1,
    const float* __restrict__ fin_b1, const float* __restrict__ fin_w2,
    const float* __restrict__ fin_b2, float* __restrict__ out) {
  __shared__ SMem s;
  const int tid  = threadIdx.x;
  const int lane = tid & 31;
  const int wave = tid >> 5;
  const int ln16 = lane & 15;
  const int hsel = lane >> 4;
  const int row0 = blockIdx.x * ROWS_PER_BLOCK + wave * 16;

  // ---- stage encoder weights into LDS (f16, unified K layout) ----
  for (int i = tid; i < 256 * KP; i += BLOCK) {
    int n = i / KP, k = i % KP;
    float w = 0.f;
    if (k >= 4 && k < 68)       w = enc_whh[n * H + (k - 4)];
    else if (k >= 68 && k < 84) w = enc_wih[n * PE2 + (k - 68)];
    s.W[i] = (_Float16)w;
  }
  for (int i = tid; i < 256; i += BLOCK) s.bias[i] = enc_bih[i] + enc_bhh[i];
  for (int i = tid; i < PE1 * INF; i += BLOCK) s.pw1[i] = pe_w1[i];
  if (tid < PE1) s.pb1[tid] = pe_b1[tid];
  for (int i = tid; i < PE2 * PE1; i += BLOCK) s.pw2[i] = pe_w2[i];
  if (tid < PE2) s.pb2[tid] = pe_b2[tid];
  for (int i = tid; i < WAVES * 16 * KP; i += BLOCK) {
    s.A[i]  = (_Float16)0.f;
    s.A2[i] = (_Float16)0.f;
  }
  __syncthreads();

  _Float16* Aw  = &s.A[wave * 16 * KP];
  _Float16* A2w = &s.A2[wave * 16 * KP];
  float* h1w = &s.h1buf[wave * 16 * PE1];
  float* f1w = &s.f1buf[wave * 16 * PD2];

  float creg[32];
#pragma unroll
  for (int i = 0; i < 32; ++i) creg[i] = 0.f;

  // =================== encoder: 50 LSTM steps ===================
  for (int t = 0; t < T; ++t) {
    {  // PE layer 1: 5 -> 32, each lane computes 16 outputs of its row
      const float* xr = x + ((long)(row0 + ln16) * T + t) * INF;
      float f0 = xr[0] * VEL_SCALING;
      float f1v = xr[1], f2 = xr[2], f3 = xr[3], f4 = xr[4];
#pragma unroll
      for (int c = 0; c < 16; ++c) {
        int col = hsel * 16 + c;
        const float* w = &s.pw1[col * INF];
        float a = s.pb1[col] + f0 * w[0] + f1v * w[1] + f2 * w[2] + f3 * w[3] + f4 * w[4];
        h1w[ln16 * PE1 + col] = fmaxf(a, 0.f);
      }
    }
    wave_sync();
    {  // PE layer 2: 32 -> 16, write f16 into A cols [68..83]
#pragma unroll
      for (int c = 0; c < 8; ++c) {
        int col = hsel * 8 + c;
        const float* w  = &s.pw2[col * PE1];
        const float* hv = &h1w[ln16 * PE1];
        float a = s.pb2[col];
#pragma unroll
        for (int k = 0; k < PE1; ++k) a += hv[k] * w[k];
        Aw[ln16 * KP + 68 + col] = (_Float16)fmaxf(a, 0.f);
      }
    }
    wave_sync();
    lstm_gates(s.W, s.bias, Aw, creg, lane);
    wave_sync();
  }

  // =================== re-stage weights for decoder ===================
  __syncthreads();
  for (int i = tid; i < 256 * KP; i += BLOCK) {
    int n = i / KP, k = i % KP;
    float w = 0.f;
    if (k < 4)       w = dec_wih[n * (2 * OUTF) + k];
    else if (k < 68) w = dec_whh[n * H + (k - 4)];
    s.W[i] = (_Float16)w;
  }
  for (int i = tid; i < 256; i += BLOCK) s.bias[i] = dec_bih[i] + dec_bhh[i];
  for (int i = tid; i < PD1 * KP; i += BLOCK) {
    int n = i / KP, k = i % KP;
    s.PDW[i] = (_Float16)((k >= 4 && k < 68) ? pd_w[n * H + (k - 4)] : 0.f);
  }
  for (int i = tid; i < PD2 * KP; i += BLOCK) {
    int n = i / KP, k = i % KP;
    s.FW1[i] = (_Float16)((k < H) ? fin_w1[n * H + k] : 0.f);
  }
  if (tid < PD1) s.pdb[tid] = pd_b[tid];
  if (tid < PD2) s.f1b[tid] = fin_b1[tid];
  if (tid < OUTF * PD2) s.fw2[tid] = fin_w2[tid];
  if (tid < OUTF) s.f2b[tid] = fin_b2[tid];
  if (tid < ROWS_PER_BLOCK) s.dsx[tid] = dataset_idx[blockIdx.x * ROWS_PER_BLOCK + tid];
  for (int i = tid; i < WAVES * 16 * KP; i += BLOCK)
    if ((i % KP) >= 68) s.A[i] = (_Float16)0.f;  // clear stale PE region
  __syncthreads();

  // head/tail state: lane holds (row = ln16, out-channel = hsel)
  float head, tail;
  {
    const float* xr = x + ((long)(row0 + ln16) * T + (T - 1)) * INF;
    float cur = xr[3 + hsel];
    float prv = xr[3 + hsel - INF];
    head = cur;
    tail = (cur - prv) * (1.0f / DTc);
  }

  // =================== decoder: 35 autoregressive steps ===================
  for (int ts = 0; ts < TS; ++ts) {
    Aw[ln16 * KP + hsel]     = (_Float16)head;  // cols 0..1
    Aw[ln16 * KP + 2 + hsel] = (_Float16)tail;  // cols 2..3
    wave_sync();
    lstm_gates(s.W, s.bias, Aw, creg, lane);
    wave_sync();
    {  // projection 64 -> 64 (+bias, relu, * per-dataset scale) -> A2 cols [0..63]
      v16h p[3];
      p[0] = load_a(Aw, lane, 0);
      p[1] = load_a(Aw, lane, 1);
      p[2] = load_a(Aw, lane, 2);
      const int nb[4] = {0, 16, 32, 48};
      v8f pdv[4];
      mm4x3_pipe(s.PDW, nb, p, pdv, lane);
#pragma unroll
      for (int j = 0; j < 4; ++j) {
        const float bP = s.pdb[16 * j + ln16];
#pragma unroll
        for (int v = 0; v < 8; ++v) {
          int m = v + 8 * hsel;
          int col = 16 * j + ln16;
          float sc = scales[s.dsx[wave * 16 + m] * PD1 + col];
          A2w[m * KP + col] = (_Float16)(fmaxf(pdv[j][v] + bP, 0.f) * sc);
        }
      }
    }
    wave_sync();
    {  // fin1: 64 -> 16 (+bias, relu) via 2 WMMA K-chunks, B preloaded
      v16h q0 = load_a(A2w, lane, 0);
      v16h q1 = load_a(A2w, lane, 1);
      v16h b0 = load_b(s.FW1, lane, 0, 0);
      v16h b1 = load_b(s.FW1, lane, 0, 1);
      v8f z = {0.f, 0.f, 0.f, 0.f, 0.f, 0.f, 0.f, 0.f};
      v8f acc = wmma_acc(q0, b0, z);
      acc = wmma_acc(q1, b1, acc);
      const float bF1 = s.f1b[ln16];
#pragma unroll
      for (int v = 0; v < 8; ++v)
        f1w[(v + 8 * hsel) * PD2 + ln16] = fmaxf(acc[v] + bF1, 0.f);
    }
    wave_sync();
    {  // fin2: 16 -> 2, tanh*lim, emit output, update head/tail
      float o = s.f2b[hsel];
      const float* fr = &f1w[ln16 * PD2];
      const float* w2 = &s.fw2[hsel * PD2];
#pragma unroll
      for (int k = 0; k < PD2; ++k) o = fmaf(fr[k], w2[k], o);
      float lim = (hsel == 0) ? JERK_LIM : STEER_RATE_LIM;
      float ot = htanh(o) * lim;
      out[((long)(row0 + ln16) * TS + ts) * OUTF + hsel] = ot;
      head = fmaf(ot, DTc, head);
      tail = ot;
    }
    wave_sync();
  }
}

extern "C" void kernel_launch(void* const* d_in, const int* in_sizes, int n_in,
                              void* d_out, int out_size, void* d_ws, size_t ws_size,
                              hipStream_t stream) {
  (void)in_sizes; (void)n_in; (void)out_size; (void)d_ws; (void)ws_size;
  const float* x        = (const float*)d_in[0];
  const int*   dsidx    = (const int*)d_in[1];
  const float* pe_w1    = (const float*)d_in[2];
  const float* pe_b1    = (const float*)d_in[3];
  const float* pe_w2    = (const float*)d_in[4];
  const float* pe_b2    = (const float*)d_in[5];
  const float* enc_wih  = (const float*)d_in[6];
  const float* enc_whh  = (const float*)d_in[7];
  const float* enc_bih  = (const float*)d_in[8];
  const float* enc_bhh  = (const float*)d_in[9];
  const float* dec_wih  = (const float*)d_in[10];
  const float* dec_whh  = (const float*)d_in[11];
  const float* dec_bih  = (const float*)d_in[12];
  const float* dec_bhh  = (const float*)d_in[13];
  const float* pd_w     = (const float*)d_in[14];
  const float* pd_b     = (const float*)d_in[15];
  const float* scales   = (const float*)d_in[16];
  const float* fin_w1   = (const float*)d_in[17];
  const float* fin_b1   = (const float*)d_in[18];
  const float* fin_w2   = (const float*)d_in[19];
  const float* fin_b2   = (const float*)d_in[20];
  float* out = (float*)d_out;

  dim3 grid(Bsz / ROWS_PER_BLOCK);
  dim3 block(BLOCK);
  hipLaunchKernelGGL(predictor_kernel, grid, block, 0, stream,
                     x, dsidx, pe_w1, pe_b1, pe_w2, pe_b2,
                     enc_wih, enc_whh, enc_bih, enc_bhh,
                     dec_wih, dec_whh, dec_bih, dec_bhh,
                     pd_w, pd_b, scales, fin_w1, fin_b1, fin_w2, fin_b2, out);
}
